// SparseWeightedAdjacency_33320356282486
// MI455X (gfx1250) — matrix operand order repaired and checked by reading, not compiled, not run on hardware
//
#include <hip/hip_runtime.h>
#include <hip/hip_bf16.h>
#include <math.h>

// ---------------------------------------------------------------------------
// CDNA5 / gfx1250: wave32, v_wmma_f32_16x16x32_f16 for all O(N^2*K) GEMMs,
// async global->LDS double-buffered staging of the adjacency matrix,
// LDS-tiled fused conv chain for s_branch, NT stores for write-once outputs.
// ---------------------------------------------------------------------------

typedef __attribute__((ext_vector_type(16))) _Float16 v16h;
typedef __attribute__((ext_vector_type(8)))  _Float16 v8h;
typedef __attribute__((ext_vector_type(8)))  float    v8f;
typedef __attribute__((ext_vector_type(4)))  int      v4i;

#define T_   8
#define N_   640
#define H_   4

// ---- WMMA ------------------------------------------------------------------
__device__ __forceinline__ v8f wmma32f16(v16h a, v16h b, v8f c) {
  return __builtin_amdgcn_wmma_f32_16x16x32_f16(false, a, false, b, (short)0, c,
                                                false, false);
}
// A fragment (16x32 f16, ISA 7.12.2): lane m = lane&15, hi = lane>>4.
// halves 0..7  = K[8*hi .. 8*hi+7]   (contiguous 16B)
// halves 8..15 = K[16+8*hi .. +7]    (contiguous 16B)
union FragU { v16h v; v8h h[2]; };
__device__ __forceinline__ v16h a_frag(const _Float16* row, int hi) {
  FragU u;
  u.h[0] = *(const v8h*)(row + 8 * hi);
  u.h[1] = *(const v8h*)(row + 16 + 8 * hi);
  return u.v;
}
__device__ __forceinline__ v16h a_frag_k16(const _Float16* row, int hi) {
  FragU u;
  u.h[0] = *(const v8h*)(row + 8 * hi);   // K<16 valid
  u.h[1] = (v8h)(_Float16)0.0f;           // K>=16 padded
  return u.v;
}
// B fragment (32x16 f16): lane n = lane&15; halves 0..7 = K[8*hi..+7], 8..15 = K[16+8*hi..+7].
__device__ __forceinline__ v16h b_frag(const _Float16* colsrc, int hi) {
  FragU u;
  u.h[0] = *(const v8h*)(colsrc + 8 * hi);
  u.h[1] = *(const v8h*)(colsrc + 16 + 8 * hi);
  return u.v;
}
__device__ __forceinline__ v16h b_frag_k16(const _Float16* colsrc, int hi) {
  FragU u;
  u.h[0] = *(const v8h*)(colsrc + 8 * hi);
  u.h[1] = (v8h)(_Float16)0.0f;
  return u.v;
}

// ---- async global -> LDS (CDNA5 ASYNCcnt path, probed with fallback) -------
#if defined(__AMDGCN__) && __has_builtin(__builtin_amdgcn_global_load_async_to_lds_b128)
#define HAVE_ASYNC_LDS 1
typedef __attribute__((address_space(1))) v4i as1_v4i;
typedef __attribute__((address_space(3))) v4i as3_v4i;
#endif

__device__ __forceinline__ void async_cp16B(const _Float16* g, _Float16* l) {
#if defined(HAVE_ASYNC_LDS)
  __builtin_amdgcn_global_load_async_to_lds_b128(
      (as1_v4i*)(void*)(const void*)g, (as3_v4i*)(void*)l, 0, 0);
#else
  *(v8h*)l = *(const v8h*)g;
#endif
}
__device__ __forceinline__ void wait_async2() {
#if defined(__AMDGCN__) && __has_builtin(__builtin_amdgcn_s_wait_asynccnt)
  __builtin_amdgcn_s_wait_asynccnt(2);
#elif defined(__AMDGCN__)
  asm volatile("s_wait_asynccnt 0x2" ::: "memory");
#endif
  asm volatile("" ::: "memory");
}
__device__ __forceinline__ void wait_async0() {
#if defined(__AMDGCN__) && __has_builtin(__builtin_amdgcn_s_wait_asynccnt)
  __builtin_amdgcn_s_wait_asynccnt(0);
#elif defined(__AMDGCN__)
  asm volatile("s_wait_asynccnt 0x0" ::: "memory");
#endif
  asm volatile("" ::: "memory");
}
// Stage one 16-row x 32-half A tile (1KB) into LDS: 2 async b128 per lane.
__device__ __forceinline__ void stage_tile(const _Float16* Gbase, int k0,
                                           _Float16* buf, int lane) {
#pragma unroll
  for (int j = 0; j < 2; ++j) {
    int c = lane + 32 * j;          // 64 chunks of 16B
    int rr = c >> 2, sg = c & 3;
    async_cp16B(Gbase + (size_t)rr * N_ + k0 + sg * 8, buf + rr * 32 + sg * 8);
  }
}

// ---------------------------------------------------------------------------
// K1: per-node prep.  grp MLP -> normalized temp (f16), se/cse x-embeddings
// (TRANSPOSED [chan][node] for contiguous B fragments), q/k split heads.
// ---------------------------------------------------------------------------
__global__ __launch_bounds__(256)
void prep_kernel(const float* __restrict__ graph,
                 const float* g_l1w, const float* g_l1b, const float* g_aP,
                 const float* g_l2w, const float* g_l2b,
                 const float* sex_w, const float* sex_b,
                 const float* csex_w, const float* csex_b,
                 const float* embw, const float* embb,
                 const float* qw, const float* qb,
                 const float* kw, const float* kb,
                 _Float16* __restrict__ tempH,
                 _Float16* __restrict__ embseT, _Float16* __restrict__ embcseT,
                 _Float16* __restrict__ qH, _Float16* __restrict__ kH) {
  int idx = blockIdx.x * 256 + threadIdx.x;   // T*N
  if (idx >= T_ * N_) return;
  int t = idx / N_, n = idx % N_;
  float s0 = graph[(size_t)idx * 3 + 1];
  float s1 = graph[(size_t)idx * 3 + 2];
  float ga = g_aP[0];
  float hb[8];
#pragma unroll
  for (int j = 0; j < 8; ++j) {
    float v = g_l1b[j] + g_l1w[j * 2] * s0 + g_l1w[j * 2 + 1] * s1;
    hb[j] = (v >= 0.f) ? v : ga * v;
  }
  float nf[16], nrm = 0.f;
#pragma unroll
  for (int c = 0; c < 16; ++c) {
    float v = g_l2b[c];
#pragma unroll
    for (int j = 0; j < 8; ++j) v += g_l2w[c * 8 + j] * hb[j];
    nf[c] = v; nrm += v * v;
  }
  float inv = 1.0f / fmaxf(sqrtf(nrm), 1e-12f);
#pragma unroll
  for (int c = 0; c < 16; ++c) tempH[(size_t)idx * 16 + c] = (_Float16)(nf[c] * inv);
#pragma unroll
  for (int c = 0; c < 16; ++c) {
    size_t io = ((size_t)t * 16 + c) * N_ + n;
    embseT [io] = (_Float16)(sex_b [c] + sex_w [c * 2] * s0 + sex_w [c * 2 + 1] * s1);
    embcseT[io] = (_Float16)(csex_b[c] + csex_w[c * 2] * s0 + csex_w[c * 2 + 1] * s1);
  }
  float emb[64];
  for (int c = 0; c < 64; ++c) emb[c] = embb[c] + embw[c * 2] * s0 + embw[c * 2 + 1] * s1;
  for (int o = 0; o < 64; ++o) {
    float qv = qb[o], kv = kb[o];
    for (int i = 0; i < 64; ++i) { qv += qw[o * 64 + i] * emb[i]; kv += kw[o * 64 + i] * emb[i]; }
    int hh = o >> 4, cc = o & 15;
    size_t io = ((((size_t)t * H_) + hh) * N_ + n) * 16 + cc;
    qH[io] = (_Float16)qv; kH[io] = (_Float16)kv;
  }
}

// ---------------------------------------------------------------------------
// K2: corr = temp @ temp^T (WMMA, K=16 padded), fused binary threshold.
// ---------------------------------------------------------------------------
__global__ __launch_bounds__(256)
void corr_kernel(const _Float16* __restrict__ tempH, const float* __restrict__ thP,
                 float* __restrict__ outS, float* __restrict__ outC,
                 _Float16* __restrict__ GH, _Float16* __restrict__ CGH) {
  int wid = threadIdx.x >> 5, lane = threadIdx.x & 31;
  int tile = blockIdx.x * 8 + wid;          // T * 40 * 40
  int t = tile / 1600, rem = tile % 1600;
  int ti = rem / 40, tj = rem % 40;
  int n16 = lane & 15, hi = lane >> 4;
  float th = fminf(fmaxf(thP[0], -0.9999f), 0.9999f);
  const _Float16* Arow = tempH + ((size_t)t * N_ + ti * 16 + n16) * 16;
  const _Float16* Brow = tempH + ((size_t)t * N_ + tj * 16 + n16) * 16;
  v8f c = {};
  c = wmma32f16(a_frag_k16(Arow, hi), b_frag_k16(Brow, hi), c);
#pragma unroll
  for (int r = 0; r < 8; ++r) {
    int m = ti * 16 + r + 8 * hi;
    int n = tj * 16 + n16;
    size_t idx = ((size_t)t * N_ + m) * N_ + n;
    float sg = (c[r] - th > 0.f) ? 1.0f : 0.0f;
    float cg = 1.0f - sg + 1e-6f;
    outS[idx] = sg; outC[idx] = cg;
    GH[idx] = (_Float16)sg; CGH[idx] = (_Float16)cg;
  }
}

// ---------------------------------------------------------------------------
// K3: row sums of S_G (G symmetric -> row sums == col sums).
// ---------------------------------------------------------------------------
__global__ __launch_bounds__(256)
void sums_kernel(const float* __restrict__ outS,
                 float* __restrict__ divS, float* __restrict__ divC) {
  __shared__ float red[256];
  int row = blockIdx.x;                     // T*N
  const float* p = outS + (size_t)row * N_;
  float s = 0.f;
  for (int i = threadIdx.x; i < N_; i += 256) s += p[i];
  red[threadIdx.x] = s; __syncthreads();
  for (int st = 128; st > 0; st >>= 1) {
    if (threadIdx.x < st) red[threadIdx.x] += red[threadIdx.x + st];
    __syncthreads();
  }
  if (threadIdx.x == 0) {
    divS[row] = red[0];
    divC[row] = (float)N_ * (1.0f + 1e-6f) - red[0];
  }
}

// ---------------------------------------------------------------------------
// K4: edge_initial (WMMA, M=640,N=16,K=640), A tiles async-staged in LDS,
// fused e/eq/ek 16x16 linears.  One wave per 16-row tile.
// ---------------------------------------------------------------------------
__global__ __launch_bounds__(32)
void edge_kernel(const _Float16* __restrict__ GH, const float* __restrict__ divR,
                 const _Float16* __restrict__ embT,
                 const float* We, const float* be,
                 const float* Weq, const float* beq,
                 const float* Wek, const float* bek,
                 float* __restrict__ outE,
                 _Float16* __restrict__ eqT, _Float16* __restrict__ ekT) {
  __shared__ _Float16 ldsA[2][16 * 32];
  __shared__ float sm[16][16];
  __shared__ float se_[16][16];
  int lane = threadIdx.x & 31;
  int blk = blockIdx.x;                     // T * 40
  int t = blk / 40, mt = blk % 40;
  int n16 = lane & 15, hi = lane >> 4;
  const _Float16* Gbase = GH + ((size_t)t * N_ + mt * 16) * (size_t)N_;
  const _Float16* Bn = embT + ((size_t)t * 16 + n16) * N_;
  v8f c = {};
  stage_tile(Gbase, 0, ldsA[0], lane);
  for (int k0 = 0; k0 < N_; k0 += 32) {
    int cur = (k0 >> 5) & 1;
    if (k0 + 32 < N_) { stage_tile(Gbase, k0 + 32, ldsA[cur ^ 1], lane); wait_async2(); }
    else wait_async0();
    v16h a = a_frag(&ldsA[cur][n16 * 32], hi);
    v16h b = b_frag(Bn + k0, hi);
    c = wmma32f16(a, b, c);
  }
#pragma unroll
  for (int r = 0; r < 8; ++r) {
    int m = r + 8 * hi;
    sm[m][n16] = c[r] / divR[t * N_ + mt * 16 + m];
  }
  __syncthreads();
#pragma unroll
  for (int r = 0; r < 8; ++r) {
    int m = r + 8 * hi;
    float e = be[n16];
#pragma unroll
    for (int d = 0; d < 16; ++d) e += We[n16 * 16 + d] * sm[m][d];
    se_[m][n16] = e;
  }
  __syncthreads();
#pragma unroll
  for (int r = 0; r < 8; ++r) {
    int m = r + 8 * hi;
    int mg = mt * 16 + m;
    float e = se_[m][n16];
    for (int hh = 0; hh < H_; ++hh)
      outE[((((size_t)t * H_ + hh) * N_) + mg) * 16 + n16] = e;
    float qv = beq[n16], kv = bek[n16];
#pragma unroll
    for (int d = 0; d < 16; ++d) {
      qv += Weq[n16 * 16 + d] * se_[m][d];
      kv += Wek[n16 * 16 + d] * se_[m][d];
    }
    eqT[((size_t)t * 16 + n16) * N_ + mg] = (_Float16)qv;
    ekT[((size_t)t * 16 + n16) * N_ + mg] = (_Float16)kv;
  }
}

// ---------------------------------------------------------------------------
// K5: Gq = q + (G@eq)/div, Gk = k + (G@ek)/div.  agg head-independent ->
// one GEMM per (t, m-tile), 4 head writes.  A async-staged via LDS.
// ---------------------------------------------------------------------------
__global__ __launch_bounds__(32)
void gqgk_kernel(const _Float16* __restrict__ GH, const float* __restrict__ divR,
                 const _Float16* __restrict__ eqT, const _Float16* __restrict__ ekT,
                 const _Float16* __restrict__ qH, const _Float16* __restrict__ kH,
                 _Float16* __restrict__ GqH, _Float16* __restrict__ GkH) {
  __shared__ _Float16 ldsA[2][16 * 32];
  int lane = threadIdx.x & 31;
  int blk = blockIdx.x;                     // T * 40
  int t = blk / 40, mt = blk % 40;
  int n16 = lane & 15, hi = lane >> 4;
  const _Float16* Gbase = GH + ((size_t)t * N_ + mt * 16) * (size_t)N_;
  const _Float16* Bq = eqT + ((size_t)t * 16 + n16) * N_;
  const _Float16* Bk = ekT + ((size_t)t * 16 + n16) * N_;
  v8f cq = {}, ck = {};
  stage_tile(Gbase, 0, ldsA[0], lane);
  for (int k0 = 0; k0 < N_; k0 += 32) {
    int cur = (k0 >> 5) & 1;
    if (k0 + 32 < N_) { stage_tile(Gbase, k0 + 32, ldsA[cur ^ 1], lane); wait_async2(); }
    else wait_async0();
    v16h a = a_frag(&ldsA[cur][n16 * 32], hi);
    cq = wmma32f16(a, b_frag(Bq + k0, hi), cq);
    ck = wmma32f16(a, b_frag(Bk + k0, hi), ck);
  }
#pragma unroll
  for (int r = 0; r < 8; ++r) {
    int m = r + 8 * hi;
    float dv = divR[t * N_ + mt * 16 + m];
    float aq = cq[r] / dv, ak = ck[r] / dv;
    for (int h4 = 0; h4 < H_; ++h4) {
      size_t io = ((((size_t)t * H_ + h4) * N_) + mt * 16 + m) * 16 + n16;
      GqH[io] = (_Float16)((float)qH[io] + aq);
      GkH[io] = (_Float16)((float)kH[io] + ak);
    }
  }
}

// ---------------------------------------------------------------------------
// K6: scores = (Gq @ Gk^T)/8  (WMMA, K=16 padded).  8 waves/block.
// ---------------------------------------------------------------------------
__global__ __launch_bounds__(256)
void scores_kernel(const _Float16* __restrict__ GqH, const _Float16* __restrict__ GkH,
                   float* __restrict__ S) {
  int wid = threadIdx.x >> 5, lane = threadIdx.x & 31;
  int tile = blockIdx.x * 8 + wid;          // T*H * 40 * 40
  int th = tile / 1600, rem = tile % 1600;
  int mi = rem / 40, nj = rem % 40;
  int n16 = lane & 15, hi = lane >> 4;
  const _Float16* Arow = GqH + (((size_t)th * N_) + mi * 16 + n16) * 16;
  const _Float16* Brow = GkH + (((size_t)th * N_) + nj * 16 + n16) * 16;
  v8f c = {};
  c = wmma32f16(a_frag_k16(Arow, hi), b_frag_k16(Brow, hi), c);
#pragma unroll
  for (int r = 0; r < 8; ++r) {
    int m = mi * 16 + r + 8 * hi;
    int n = nj * 16 + n16;
    S[((size_t)th * N_ + m) * N_ + n] = c[r] * 0.125f;
  }
}

// ---------------------------------------------------------------------------
// K7: row softmax in place.
// ---------------------------------------------------------------------------
__global__ __launch_bounds__(256)
void softmax_kernel(float* __restrict__ S) {
  __shared__ float red[256];
  float* p = S + (size_t)blockIdx.x * N_;   // T*H*N rows
  int tid = threadIdx.x;
  float mx = -1e30f;
  for (int i = tid; i < N_; i += 256) mx = fmaxf(mx, p[i]);
  red[tid] = mx; __syncthreads();
  for (int s = 128; s > 0; s >>= 1) {
    if (tid < s) red[tid] = fmaxf(red[tid], red[tid + s]);
    __syncthreads();
  }
  mx = red[0]; __syncthreads();
  float sum = 0.f;
  for (int i = tid; i < N_; i += 256) sum += __expf(p[i] - mx);
  red[tid] = sum; __syncthreads();
  for (int s = 128; s > 0; s >>= 1) {
    if (tid < s) red[tid] += red[tid + s];
    __syncthreads();
  }
  float inv = 1.0f / red[0];
  for (int i = tid; i < N_; i += 256) p[i] = __expf(p[i] - mx) * inv;
}

// ---------------------------------------------------------------------------
// K8: tcn — pointwise 8x8 mix across T, prelu + residual, in place.
// ---------------------------------------------------------------------------
__global__ __launch_bounds__(256)
void tcn_kernel(float* __restrict__ S, const float* __restrict__ W,
                const float* __restrict__ b, const float* __restrict__ aP) {
  size_t pix = (size_t)blockIdx.x * 256 + threadIdx.x;   // H*N*N
  if (pix >= (size_t)H_ * N_ * N_) return;
  const size_t stride = (size_t)H_ * N_ * N_;
  float alpha = aP[0];
  float v[T_];
#pragma unroll
  for (int t = 0; t < T_; ++t) v[t] = S[t * stride + pix];
#pragma unroll
  for (int i = 0; i < T_; ++i) {
    float o = b[i];
#pragma unroll
    for (int j = 0; j < T_; ++j) o += W[i * T_ + j] * v[j];
    o = (o >= 0.f) ? o : alpha * o;
    S[i * stride + pix] = o + v[i];
  }
}

// ---------------------------------------------------------------------------
// K9: fused s_branch conv chain.  16x16 interior, halo 5, five 4-ch feature
// levels in LDS (54 KB).  Final 20->4 conv + sigmoid threshold + identity,
// NT-stored straight to d_out.
// ---------------------------------------------------------------------------
struct AsymW { const float* c1w; const float* c2w; const float* c2b;
               const float* aP;  const float* scw; };
struct SbW { AsymW ac[5]; };

#define SB_TILE 16
#define SB_HALO 5
#define SB_LT   26

__global__ __launch_bounds__(256)
void sbranch_kernel(const float* __restrict__ S, const float* __restrict__ iden,
                    float* __restrict__ outp, SbW w) {
  __shared__ float lv[5][4][SB_LT * SB_LT];
  int blk = blockIdx.x;                     // T * 40 * 40
  int t = blk / 1600, rem = blk % 1600;
  int ty = rem / 40, tx = rem % 40;
  int y0 = ty * SB_TILE - SB_HALO, x0 = tx * SB_TILE - SB_HALO;

  for (int i = threadIdx.x; i < 4 * SB_LT * SB_LT; i += 256) {
    int c = i / (SB_LT * SB_LT);
    int r = i % (SB_LT * SB_LT);
    int yy = y0 + r / SB_LT, xx = x0 + r % SB_LT;
    float v = 0.f;
    if (yy >= 0 && yy < N_ && xx >= 0 && xx < N_)
      v = S[(((size_t)t * H_ + c) * N_ + yy) * N_ + xx];
    lv[0][c][r] = v;
  }
  __syncthreads();

  for (int l = 1; l <= 4; ++l) {
    const AsymW aw = w.ac[l - 1];
    float alpha = aw.aP[0];
    int lo = l, wdt = SB_LT - 2 * l;
    int cnt = 4 * wdt * wdt;
    for (int i = threadIdx.x; i < cnt; i += 256) {
      int o = i / (wdt * wdt);
      int r = i % (wdt * wdt);
      int yy = lo + r / wdt, xx = lo + r % wdt;
      int p0 = yy * SB_LT + xx;
      float acc = aw.c2b[o];
#pragma unroll
      for (int ci = 0; ci < 4; ++ci) {
        const float* base = &lv[l - 1][ci][0];
        float xc = base[p0];
        float xm = base[p0 - SB_LT], xp = base[p0 + SB_LT];
        float xl = base[p0 - 1],     xr = base[p0 + 1];
        int wi = (o * 4 + ci) * 3;
        acc += aw.c1w[wi] * xm + aw.c1w[wi + 1] * xc + aw.c1w[wi + 2] * xp;
        acc += aw.c2w[wi] * xl + aw.c2w[wi + 1] * xc + aw.c2w[wi + 2] * xr;
      }
      float pr = (acc >= 0.f) ? acc : alpha * acc;
      lv[l][o][p0] = pr + lv[l - 1][o][p0];
    }
    __syncthreads();
  }

  const AsymW fw = w.ac[4];
  float alpha = fw.aP[0];
  for (int i = threadIdx.x; i < 4 * SB_TILE * SB_TILE; i += 256) {
    int o = i / (SB_TILE * SB_TILE);
    int r = i % (SB_TILE * SB_TILE);
    int ly = SB_HALO + r / SB_TILE, lx = SB_HALO + r % SB_TILE;
    int p0 = ly * SB_LT + lx;
    float acc = fw.c2b[o];
    float sc = 0.f;
    for (int lvI = 0; lvI < 5; ++lvI) {
#pragma unroll
      for (int ci4 = 0; ci4 < 4; ++ci4) {
        int ci = lvI * 4 + ci4;
        const float* base = &lv[lvI][ci4][0];
        float xc = base[p0];
        float xm = base[p0 - SB_LT], xp = base[p0 + SB_LT];
        float xl = base[p0 - 1],     xr = base[p0 + 1];
        int wi = (o * 20 + ci) * 3;
        acc += fw.c1w[wi] * xm + fw.c1w[wi + 1] * xc + fw.c1w[wi + 2] * xp;
        acc += fw.c2w[wi] * xl + fw.c2w[wi + 1] * xc + fw.c2w[wi + 2] * xr;
        sc  += fw.scw[o * 20 + ci] * xc;
      }
    }
    float y4 = ((acc >= 0.f) ? acc : alpha * acc) + sc;
    float thr = 1.0f / (1.0f + __expf(-y4));
    float xv = lv[0][o][p0];
    float res = (xv > thr) ? xv : 0.f;
    int gy = ty * SB_TILE + r / SB_TILE, gx = tx * SB_TILE + r % SB_TILE;
    res += iden[((size_t)t * N_ + gy) * N_ + gx];
    __builtin_nontemporal_store(res,
        &outp[(((size_t)t * H_ + o) * N_ + gy) * N_ + gx]);
  }
}

// ---------------------------------------------------------------------------
// Host launch.
// ---------------------------------------------------------------------------
extern "C" void kernel_launch(void* const* d_in, const int* in_sizes, int n_in,
                              void* d_out, int out_size, void* d_ws, size_t ws_size,
                              hipStream_t stream) {
  (void)in_sizes; (void)n_in; (void)out_size; (void)ws_size;
  auto F = [&](int i) { return (const float*)d_in[i]; };
  const float* graph    = F(0);
  const float* identity = F(1);
  // tree-flattened params (alphabetical dict keys):
  // attn: 2 cek.b 3 cek.w 4 ceq.b 5 ceq.w 6 ek.b 7 ek.w 8 emb.b 9 emb.w
  //       10 eq.b 11 eq.w 12 k.b 13 k.w 14 q.b 15 q.w
  // csb : 16..36 (ac0..ac4), 37 tcn.b 38 tcn.w 39 tcn_a
  // cse : 40 e.b 41 e.w 42 x.b 43 x.w
  // grp : 44 a 45 l1.b 46 l1.w 47 l2.b 48 l2.w 49 th
  // sb  : 50..70 (ac0..ac4), 71 tcn.b 72 tcn.w 73 tcn_a
  // se  : 74 e.b 75 e.w 76 x.b 77 x.w

  float* out = (float*)d_out;
  float* outGS  = out;
  float* outCGS = out + 13107200;
  float* outSG  = out + 26214400;
  float* outCSG = out + 29491200;
  float* outSE  = out + 32768000;
  float* outCSE = out + 33095680;

  char* ws = (char*)d_ws;
  size_t off = 0;
  auto alloc = [&](size_t bytes) -> char* {
    char* p = ws + off;
    off += (bytes + 255) & ~(size_t)255;
    return p;
  };
  _Float16* tempH   = (_Float16*)alloc((size_t)T_ * N_ * 16 * 2);
  _Float16* embseT  = (_Float16*)alloc((size_t)T_ * N_ * 16 * 2);
  _Float16* embcseT = (_Float16*)alloc((size_t)T_ * N_ * 16 * 2);
  _Float16* qH      = (_Float16*)alloc((size_t)T_ * H_ * N_ * 16 * 2);
  _Float16* kH      = (_Float16*)alloc((size_t)T_ * H_ * N_ * 16 * 2);
  _Float16* GH      = (_Float16*)alloc((size_t)T_ * N_ * N_ * 2);
  _Float16* CGH     = (_Float16*)alloc((size_t)T_ * N_ * N_ * 2);
  float*    divS    = (float*)alloc((size_t)T_ * N_ * 4);
  float*    divC    = (float*)alloc((size_t)T_ * N_ * 4);
  _Float16* eqT     = (_Float16*)alloc((size_t)T_ * N_ * 16 * 2);
  _Float16* ekT     = (_Float16*)alloc((size_t)T_ * N_ * 16 * 2);
  _Float16* GqH     = (_Float16*)alloc((size_t)T_ * H_ * N_ * 16 * 2);
  _Float16* GkH     = (_Float16*)alloc((size_t)T_ * H_ * N_ * 16 * 2);
  float*    Sbuf    = (float*)alloc((size_t)T_ * H_ * N_ * N_ * 4);

  prep_kernel<<<(T_ * N_ + 255) / 256, 256, 0, stream>>>(
      graph,
      F(46), F(45), F(44), F(48), F(47),
      F(77), F(76), F(43), F(42),
      F(9), F(8), F(15), F(14), F(13), F(12),
      tempH, embseT, embcseT, qH, kH);

  corr_kernel<<<T_ * 1600 / 8, 256, 0, stream>>>(tempH, F(49), outSG, outCSG, GH, CGH);
  sums_kernel<<<T_ * N_, 256, 0, stream>>>(outSG, divS, divC);

  for (int br = 0; br < 2; ++br) {
    const _Float16* Gp   = br ? CGH : GH;
    const float*    divp = br ? divC : divS;
    const _Float16* embp = br ? embcseT : embseT;
    const float *We  = br ? F(41) : F(75), *be  = br ? F(40) : F(74);
    const float *Weq = br ? F(5)  : F(11), *beq = br ? F(4)  : F(10);
    const float *Wek = br ? F(3)  : F(7),  *bek = br ? F(2)  : F(6);
    float* outE = br ? outCSE : outSE;
    float* outM = br ? outCGS : outGS;
    int pb = br ? 16 : 50;
    const float* tcnb = F(pb + 21);
    const float* tcnw = F(pb + 22);
    const float* tcna = F(pb + 23);
    SbW sw;
    for (int l = 0; l < 4; ++l) {
      sw.ac[l].aP  = F(pb + l * 4 + 0);
      sw.ac[l].c1w = F(pb + l * 4 + 1);
      sw.ac[l].c2b = F(pb + l * 4 + 2);
      sw.ac[l].c2w = F(pb + l * 4 + 3);
      sw.ac[l].scw = F(pb + l * 4 + 1);
    }
    sw.ac[4].aP  = F(pb + 16);
    sw.ac[4].c1w = F(pb + 17);
    sw.ac[4].c2b = F(pb + 18);
    sw.ac[4].c2w = F(pb + 19);
    sw.ac[4].scw = F(pb + 20);

    edge_kernel<<<T_ * 40, 32, 0, stream>>>(Gp, divp, embp, We, be, Weq, beq,
                                            Wek, bek, outE, eqT, ekT);
    gqgk_kernel<<<T_ * 40, 32, 0, stream>>>(Gp, divp, eqT, ekT, qH, kH, GqH, GkH);
    scores_kernel<<<T_ * H_ * 1600 / 8, 256, 0, stream>>>(GqH, GkH, Sbuf);
    softmax_kernel<<<T_ * H_ * N_, 256, 0, stream>>>(Sbuf);
    tcn_kernel<<<(H_ * N_ * N_ + 255) / 256, 256, 0, stream>>>(Sbuf, tcnw, tcnb, tcna);
    sbranch_kernel<<<T_ * 1600, 256, 0, stream>>>(Sbuf, identity, outM, sw);
  }
}
